// BiMamba2_78305843740956
// MI455X (gfx1250) — matrix operand order repaired and checked by reading, not compiled, not run on hardware
//
#include <hip/hip_runtime.h>
#include <math.h>

// BiMamba2 on gfx1250 (MI455X): all contractions via v_wmma_f32_16x16x32_bf16 (wave32),
// GEMM staging via GLOBAL_LOAD_ASYNC_TO_LDS_B128 when the toolchain exposes it.
// Workspace requirement: ~500 MB of scratch.

typedef __bf16 bf16;
typedef __attribute__((ext_vector_type(16))) __bf16 v16bf;
typedef __attribute__((ext_vector_type(8)))  float  v8f;
typedef int v4i __attribute__((vector_size(16)));
typedef __attribute__((address_space(1))) v4i global_v4i;
typedef __attribute__((address_space(3))) v4i local_v4i;

#define LB 4096          // sequence length
#define NBATCH 2
#define DINNER 2048
#define NPROJ 4256
#define CONVDIM 2176
#define CHUNKL 256

#if defined(__gfx1250__) && __has_builtin(__builtin_amdgcn_global_load_async_to_lds_b128) && \
    __has_builtin(__builtin_amdgcn_s_wait_asynccnt)
#define USE_ASYNC 1
#else
#define USE_ASYNC 0
#endif

__device__ __forceinline__ bf16 f2bf(float x) {
  unsigned u = __builtin_bit_cast(unsigned, x);
  unsigned r = (u + 0x7fffu + ((u >> 16) & 1u)) >> 16;
  unsigned short s = (unsigned short)r;
  return __builtin_bit_cast(bf16, s);
}
__device__ __forceinline__ v8f zero8() {
  v8f z = {0.f, 0.f, 0.f, 0.f, 0.f, 0.f, 0.f, 0.f};
  return z;
}
__device__ __forceinline__ int glidx(int dir, int l) { return dir ? (LB - 1 - l) : l; }

// 16-byte copy global->LDS: async (ASYNCcnt) when available, else through VGPRs.
__device__ __forceinline__ void cp16(bf16* dst, const bf16* src) {
#if USE_ASYNC
  __builtin_amdgcn_global_load_async_to_lds_b128((global_v4i*)src, (local_v4i*)dst, 0, 0);
#else
  *(uint4*)dst = *(const uint4*)src;
#endif
}
__device__ __forceinline__ void cp_wait() {
#if USE_ASYNC
  __builtin_amdgcn_s_wait_asynccnt(0);
#endif
}

// A fragment (16x32 bf16, M x K). Lanes 0-15: M=lane, K=0..15;
// lanes 16-31: M=lane-16, K=16..31. Tile is row-major [m][k], stride ldw.
__device__ __forceinline__ v16bf load_frag_a(const bf16* __restrict__ t, int ldw, int mBase, int lane) {
  int m = mBase + (lane & 15);
  int hi = lane >> 4;
  const bf16* p = t + m * ldw + hi * 16;
  v16bf f;
#pragma unroll
  for (int i = 0; i < 16; ++i) f[i] = p[i];
  return f;
}
// B fragment (32x16 bf16, K x N). Lanes 0-15: N=lane, K=0..15; lanes 16-31: N=lane-16, K=16..31.
// Tile stored N-major [n][k] so each lane reads contiguous K.
__device__ __forceinline__ v16bf load_frag_b(const bf16* __restrict__ t, int ldw, int nBase, int lane) {
  int n = nBase + (lane & 15);
  int hi = lane >> 4;
  const bf16* p = t + n * ldw + hi * 16;
  v16bf f;
#pragma unroll
  for (int i = 0; i < 16; ++i) f[i] = p[i];
  return f;
}
__device__ __forceinline__ v8f wmma_bf16(v16bf a, v16bf b, v8f c) {
  return __builtin_amdgcn_wmma_f32_16x16x32_bf16(false, a, false, b, (short)0, c, false, false);
}
// C/D f32 16x16: VGPR r, lanes 0-15 -> (M=r, N=lane); lanes 16-31 -> (M=r+8, N=lane-16).
__device__ __forceinline__ void store_frag(float* __restrict__ C, size_t ldc, int mBase, int nBase,
                                           int lane, v8f acc, int M, int N) {
  int n = nBase + (lane & 15);
  int mo = (lane >> 4) * 8;
  if (n >= N) return;
#pragma unroll
  for (int r = 0; r < 8; ++r) {
    int m = mBase + mo + r;
    if (m < M) C[(size_t)m * ldc + n] = acc[r];
  }
}

// ---------------- f32 -> bf16 conversion (one-time, so GEMM staging is a pure copy) ----------
__global__ void cvt_bf16_kernel(const float* __restrict__ src, bf16* __restrict__ dst, int n4) {
  int i = blockIdx.x * blockDim.x + threadIdx.x;
  if (i >= n4) return;
  float4 v = ((const float4*)src)[i];
  bf16* d = dst + (size_t)i * 4;
  d[0] = f2bf(v.x); d[1] = f2bf(v.y); d[2] = f2bf(v.z); d[3] = f2bf(v.w);
}

// ---------------- GEMM: C[M,N] = A[M,K] * W[N,K]^T, bf16 inputs, f32 accumulate --------------
// Block tile 128x256, 8 waves, each wave 64x64 (16 WMMA accumulators).
__global__ __launch_bounds__(256) void gemm_bf16_kernel(const bf16* __restrict__ A,
                                                        const bf16* __restrict__ W,
                                                        float* __restrict__ C, int M, int N, int K) {
  __shared__ __align__(16) bf16 As[128 * 40];
  __shared__ __align__(16) bf16 Bs[256 * 40];
  const int tid = threadIdx.x, lane = tid & 31, wv = tid >> 5;
  const int mBase = blockIdx.x * 128, nBase = blockIdx.y * 256;
  const int mW = (wv >> 2) * 64, nW = (wv & 3) * 64;
  v8f acc[4][4];
#pragma unroll
  for (int i = 0; i < 4; ++i)
#pragma unroll
    for (int j = 0; j < 4; ++j) acc[i][j] = zero8();

  for (int k0 = 0; k0 < K; k0 += 32) {
    // stage A tile: 128 rows x 32 bf16 = 512 x 16B
    for (int idx = tid; idx < 512; idx += 256) {
      int r = idx >> 2, c = (idx & 3) * 8;
      cp16(As + r * 40 + c, A + (size_t)(mBase + r) * K + k0 + c);
    }
    // stage W tile: 256 rows x 32 bf16 = 1024 x 16B (clamp OOB rows; store is guarded)
    for (int idx = tid; idx < 1024; idx += 256) {
      int r = idx >> 2, c = (idx & 3) * 8;
      int gn = nBase + r;
      if (gn >= N) gn = N - 1;
      cp16(Bs + r * 40 + c, W + (size_t)gn * K + k0 + c);
    }
    cp_wait();
    __syncthreads();
    v16bf fa[4];
#pragma unroll
    for (int i = 0; i < 4; ++i) fa[i] = load_frag_a(As, 40, mW + 16 * i, lane);
#pragma unroll
    for (int j = 0; j < 4; ++j) {
      v16bf fb = load_frag_b(Bs, 40, nW + 16 * j, lane);
#pragma unroll
      for (int i = 0; i < 4; ++i) acc[i][j] = wmma_bf16(fa[i], fb, acc[i][j]);
    }
    __syncthreads();
  }
#pragma unroll
  for (int i = 0; i < 4; ++i)
#pragma unroll
    for (int j = 0; j < 4; ++j)
      store_frag(C, N, mBase + mW + 16 * i, nBase + nW + 16 * j, lane, acc[i][j], M, N);
}

// ---------------- dt = softplus(raw + bias); a = dt * (-exp(A_log)) --------------------------
__global__ void dt_kernel(const float* __restrict__ zx, const float* __restrict__ dt_bias,
                          const float* __restrict__ A_log, float* __restrict__ dtb,
                          float* __restrict__ ab, int total) {
  int tid = blockIdx.x * blockDim.x + threadIdx.x;
  if (tid >= total) return;
  int h = tid & 31;
  int bl = tid >> 5;
  float v = zx[(size_t)bl * NPROJ + 4224 + h] + dt_bias[h];
  float dt = (v > 20.f) ? v : log1pf(expf(v));
  dtb[tid] = dt;
  ab[tid] = dt * (-expf(A_log[h]));
}

// ---------------- depthwise conv7 + SiLU; scatter to Xdt (x*dt), B, C ------------------------
__global__ void conv_kernel(const float* __restrict__ zx, const float* __restrict__ cw,
                            const float* __restrict__ cbias, const float* __restrict__ dtb,
                            float* __restrict__ Xdt, float* __restrict__ Bm,
                            float* __restrict__ Cm, int total) {
  int tid = blockIdx.x * blockDim.x + threadIdx.x;
  if (tid >= total) return;
  int ch = tid % CONVDIM;
  int bl = tid / CONVDIM;
  int l = bl & (LB - 1);
  float acc = cbias[ch];
#pragma unroll
  for (int j = 0; j < 7; ++j) {
    int ll = l + j - 3;
    if (ll >= 0 && ll < LB)
      acc += cw[ch * 7 + j] * zx[((size_t)(bl + ll - l)) * NPROJ + DINNER + ch];
  }
  float s = acc / (1.f + expf(-acc));
  if (ch < DINNER) {
    int h = ch >> 6;
    Xdt[(size_t)bl * DINNER + ch] = s * dtb[bl * 32 + h];
  } else if (ch < DINNER + 64) {
    Bm[(size_t)bl * 64 + (ch - DINNER)] = s;
  } else {
    Cm[(size_t)bl * 64 + (ch - DINNER - 64)] = s;
  }
}

// ---------------- per-(b,dir,chunk,head) cumsum of a; chunk totals ---------------------------
__global__ void cumsum_kernel(const float* __restrict__ ab, float* __restrict__ Acum,
                              float* __restrict__ tb) {
  int t = blockIdx.x * blockDim.x + threadIdx.x;
  if (t >= 2048) return;
  int h = t & 31, c = (t >> 5) & 15, dir = (t >> 9) & 1, b = t >> 10;
  float run = 0.f;
  for (int i = 0; i < CHUNKL; ++i) {
    int lg = glidx(dir, c * CHUNKL + i);
    run += ab[((size_t)(b * LB + lg)) * 32 + h];
    Acum[(size_t)t * CHUNKL + i] = run;
  }
  tb[t] = run;
}

// ---------------- G[l,s] = sum_n C[l,n]*B[s,n], per (b,dir,chunk), WMMA ----------------------
__global__ __launch_bounds__(128) void ssd_g_kernel(const float* __restrict__ Bm,
                                                    const float* __restrict__ Cm,
                                                    float* __restrict__ G) {
  __shared__ bf16 As[64 * 40];
  __shared__ bf16 Bs[64 * 40];
  const int tid = threadIdx.x, lane = tid & 31, wv = tid >> 5;
  const int z = blockIdx.z;
  const int c = z & 15, dir = (z >> 4) & 1, b = z >> 5;
  const int mBase = blockIdx.x * 64, nBase = blockIdx.y * 64;
  const int mW = (wv >> 1) * 32, nW = (wv & 1) * 32;
  v8f a00 = zero8(), a01 = zero8(), a10 = zero8(), a11 = zero8();
  for (int n0 = 0; n0 < 64; n0 += 32) {
    for (int idx = tid; idx < 2048; idx += 128) {
      int r = idx >> 5, k = idx & 31;
      int lg = glidx(dir, c * CHUNKL + mBase + r);
      As[r * 40 + k] = f2bf(Cm[((size_t)(b * LB + lg)) * 64 + n0 + k]);
    }
    for (int idx = tid; idx < 2048; idx += 128) {
      int r = idx >> 5, k = idx & 31;
      int lg = glidx(dir, c * CHUNKL + nBase + r);
      Bs[r * 40 + k] = f2bf(Bm[((size_t)(b * LB + lg)) * 64 + n0 + k]);
    }
    __syncthreads();
    v16bf fa0 = load_frag_a(As, 40, mW, lane);
    v16bf fa1 = load_frag_a(As, 40, mW + 16, lane);
    v16bf fb0 = load_frag_b(Bs, 40, nW, lane);
    v16bf fb1 = load_frag_b(Bs, 40, nW + 16, lane);
    a00 = wmma_bf16(fa0, fb0, a00);
    a01 = wmma_bf16(fa0, fb1, a01);
    a10 = wmma_bf16(fa1, fb0, a10);
    a11 = wmma_bf16(fa1, fb1, a11);
    __syncthreads();
  }
  float* Gz = G + (size_t)z * 65536;
  store_frag(Gz, 256, mBase + mW,      nBase + nW,      lane, a00, 256, 256);
  store_frag(Gz, 256, mBase + mW,      nBase + nW + 16, lane, a01, 256, 256);
  store_frag(Gz, 256, mBase + mW + 16, nBase + nW,      lane, a10, 256, 256);
  store_frag(Gz, 256, mBase + mW + 16, nBase + nW + 16, lane, a11, 256, 256);
}

// ---------------- states[p,n] = sum_l exp(t - Acum[l]) * X[l,p] * B[l,n], WMMA ---------------
__global__ __launch_bounds__(128) void ssd_states_kernel(const float* __restrict__ Xdt,
                                                         const float* __restrict__ Bm,
                                                         const float* __restrict__ Acum,
                                                         const float* __restrict__ tb,
                                                         float* __restrict__ states) {
  __shared__ bf16 As[64 * 40];
  __shared__ bf16 Bs[64 * 40];
  const int tid = threadIdx.x, lane = tid & 31, wv = tid >> 5;
  const int t = blockIdx.x;
  const int h = t & 31, c = (t >> 5) & 15, dir = (t >> 9) & 1, b = t >> 10;
  const int mW = (wv >> 1) * 32, nW = (wv & 1) * 32;
  const float tv = tb[t];
  v8f a00 = zero8(), a01 = zero8(), a10 = zero8(), a11 = zero8();
  for (int l0 = 0; l0 < CHUNKL; l0 += 32) {
    for (int idx = tid; idx < 2048; idx += 128) {
      int p = idx & 63, i = idx >> 6;
      int lg = glidx(dir, c * CHUNKL + l0 + i);
      float dec = expf(tv - Acum[(size_t)t * CHUNKL + l0 + i]);
      As[p * 40 + i] = f2bf(Xdt[((size_t)(b * LB + lg)) * DINNER + h * 64 + p] * dec);
    }
    for (int idx = tid; idx < 2048; idx += 128) {
      int n = idx & 63, i = idx >> 6;
      int lg = glidx(dir, c * CHUNKL + l0 + i);
      Bs[n * 40 + i] = f2bf(Bm[((size_t)(b * LB + lg)) * 64 + n]);
    }
    __syncthreads();
    v16bf fa0 = load_frag_a(As, 40, mW, lane);
    v16bf fa1 = load_frag_a(As, 40, mW + 16, lane);
    v16bf fb0 = load_frag_b(Bs, 40, nW, lane);
    v16bf fb1 = load_frag_b(Bs, 40, nW + 16, lane);
    a00 = wmma_bf16(fa0, fb0, a00);
    a01 = wmma_bf16(fa0, fb1, a01);
    a10 = wmma_bf16(fa1, fb0, a10);
    a11 = wmma_bf16(fa1, fb1, a11);
    __syncthreads();
  }
  float* S = states + (size_t)t * 4096;
  store_frag(S, 64, mW,      nW,      lane, a00, 64, 64);
  store_frag(S, 64, mW,      nW + 16, lane, a01, 64, 64);
  store_frag(S, 64, mW + 16, nW,      lane, a10, 64, 64);
  store_frag(S, 64, mW + 16, nW + 16, lane, a11, 64, 64);
}

// ---------------- inter-chunk scan: prev_c = exp(t_{c-1})*prev_{c-1} + states_{c-1} ----------
__global__ void scan_kernel(const float* __restrict__ states, const float* __restrict__ tb,
                            float* __restrict__ prev) {
  int blk = blockIdx.x;  // (b,dir,h)
  int h = blk & 31, dir = (blk >> 5) & 1, b = blk >> 6;
  int tid = threadIdx.x;
  float P[16];
#pragma unroll
  for (int j = 0; j < 16; ++j) P[j] = 0.f;
  for (int c = 0; c < 16; ++c) {
    int t = ((b * 2 + dir) * 16 + c) * 32 + h;
    size_t off = (size_t)t * 4096;
    float sc = expf(tb[t]);
#pragma unroll
    for (int j = 0; j < 16; ++j) {
      int e = j * 256 + tid;
      prev[off + e] = P[j];
      P[j] = sc * P[j] + states[off + e];
    }
  }
}

// ---------------- Y = (G . Ldec) @ X + (exp(Acum).C) @ prev^T, WMMA, scatter to orig l -------
__global__ __launch_bounds__(256) void ssd_y_kernel(const float* __restrict__ G,
                                                    const float* __restrict__ Xdt,
                                                    const float* __restrict__ Cm,
                                                    const float* __restrict__ prev,
                                                    const float* __restrict__ Acum,
                                                    float* __restrict__ yf,
                                                    float* __restrict__ yb) {
  __shared__ bf16 Gs[256 * 40];
  __shared__ bf16 Xs[64 * 40];
  const int tid = threadIdx.x, lane = tid & 31, wv = tid >> 5;
  const int t = blockIdx.x;
  const int h = t & 31, c = (t >> 5) & 15, dir = (t >> 9) & 1, b = t >> 10;
  const int z = t >> 5;
  v8f acc[2][4];
#pragma unroll
  for (int i = 0; i < 2; ++i)
#pragma unroll
    for (int j = 0; j < 4; ++j) acc[i][j] = zero8();
  const float acl = Acum[(size_t)t * CHUNKL + tid];  // this thread's row l = tid
  // Phase 1: diagonal block  Y += (G*L) @ X
  for (int s0 = 0; s0 < CHUNKL; s0 += 32) {
    {
      const float* Gr = G + (size_t)z * 65536 + (size_t)tid * 256;
      bf16* d = Gs + tid * 40;
      for (int s = 0; s < 32; ++s) {
        int sg = s0 + s;
        float dec = (sg <= tid) ? expf(acl - Acum[(size_t)t * CHUNKL + sg]) : 0.f;
        d[s] = f2bf(Gr[sg] * dec);
      }
    }
    for (int idx = tid; idx < 2048; idx += 256) {
      int p = idx & 63, i = idx >> 6;
      int lg = glidx(dir, c * CHUNKL + s0 + i);
      Xs[p * 40 + i] = f2bf(Xdt[((size_t)(b * LB + lg)) * DINNER + h * 64 + p]);
    }
    __syncthreads();
    v16bf fa0 = load_frag_a(Gs, 40, wv * 32, lane);
    v16bf fa1 = load_frag_a(Gs, 40, wv * 32 + 16, lane);
#pragma unroll
    for (int j = 0; j < 4; ++j) {
      v16bf fb = load_frag_b(Xs, 40, j * 16, lane);
      acc[0][j] = wmma_bf16(fa0, fb, acc[0][j]);
      acc[1][j] = wmma_bf16(fa1, fb, acc[1][j]);
    }
    __syncthreads();
  }
  // Phase 2: off-diagonal  Y += (exp(Acum) * C) @ prev^T
  const float eal = expf(acl);
  for (int n0 = 0; n0 < 64; n0 += 32) {
    {
      int lg = glidx(dir, c * CHUNKL + tid);
      const float* Cr = Cm + ((size_t)(b * LB + lg)) * 64 + n0;
      bf16* d = Gs + tid * 40;
      for (int k = 0; k < 32; ++k) d[k] = f2bf(Cr[k] * eal);
    }
    for (int idx = tid; idx < 2048; idx += 256) {
      int p = idx & 63, i = idx >> 6;
      Xs[p * 40 + i] = f2bf(prev[(size_t)t * 4096 + p * 64 + n0 + i]);
    }
    __syncthreads();
    v16bf fa0 = load_frag_a(Gs, 40, wv * 32, lane);
    v16bf fa1 = load_frag_a(Gs, 40, wv * 32 + 16, lane);
#pragma unroll
    for (int j = 0; j < 4; ++j) {
      v16bf fb = load_frag_b(Xs, 40, j * 16, lane);
      acc[0][j] = wmma_bf16(fa0, fb, acc[0][j]);
      acc[1][j] = wmma_bf16(fa1, fb, acc[1][j]);
    }
    __syncthreads();
  }
  float* dst = dir ? yb : yf;
#pragma unroll
  for (int i = 0; i < 2; ++i)
#pragma unroll
    for (int j = 0; j < 4; ++j) {
      int col = h * 64 + j * 16 + (lane & 15);
      int mo = (lane >> 4) * 8;
#pragma unroll
      for (int r = 0; r < 8; ++r) {
        int l = wv * 32 + i * 16 + mo + r;
        int lg = glidx(dir, c * CHUNKL + l);
        dst[((size_t)(b * LB + lg)) * DINNER + col] = acc[i][j][r];
      }
    }
}

// ------- y=(yf+yb-diag); RMS norm; gate with silu(z); emit bf16 for the out_proj GEMM --------
__global__ __launch_bounds__(256) void norm_kernel(const float* __restrict__ yf,
                                                   const float* __restrict__ yb,
                                                   const float* __restrict__ Xdt,
                                                   const float* __restrict__ Bm,
                                                   const float* __restrict__ Cm,
                                                   const float* __restrict__ zx,
                                                   const float* __restrict__ nw,
                                                   bf16* __restrict__ ynbf) {
  __shared__ float red[256];
  __shared__ float sh0;
  int bl = blockIdx.x;
  int tid = threadIdx.x;
  float v = 0.f;
  if (tid < 64) v = Cm[(size_t)bl * 64 + tid] * Bm[(size_t)bl * 64 + tid];
  red[tid] = v;
  __syncthreads();
  for (int s = 128; s > 0; s >>= 1) {
    if (tid < s) red[tid] += red[tid + s];
    __syncthreads();
  }
  if (tid == 0) sh0 = red[0];
  __syncthreads();
  float cbv = sh0;
  float yv[8];
  float ss = 0.f;
#pragma unroll
  for (int j = 0; j < 8; ++j) {
    int ch = j * 256 + tid;
    size_t idx = (size_t)bl * DINNER + ch;
    float y = yf[idx] + yb[idx] - cbv * Xdt[idx];
    yv[j] = y;
    ss += y * y;
  }
  __syncthreads();
  red[tid] = ss;
  __syncthreads();
  for (int s = 128; s > 0; s >>= 1) {
    if (tid < s) red[tid] += red[tid + s];
    __syncthreads();
  }
  if (tid == 0) sh0 = rsqrtf(red[0] / (float)DINNER + 1e-5f);
  __syncthreads();
  float rstd = sh0;
#pragma unroll
  for (int j = 0; j < 8; ++j) {
    int ch = j * 256 + tid;
    size_t idx = (size_t)bl * DINNER + ch;
    float zg = zx[(size_t)bl * NPROJ + ch];
    ynbf[idx] = f2bf(yv[j] * rstd * nw[ch] * (zg / (1.f + expf(-zg))));
  }
}

extern "C" void kernel_launch(void* const* d_in, const int* in_sizes, int n_in,
                              void* d_out, int out_size, void* d_ws, size_t ws_size,
                              hipStream_t stream) {
  (void)in_sizes; (void)n_in; (void)out_size; (void)ws_size;
  const float* u      = (const float*)d_in[0];
  const float* ipw    = (const float*)d_in[1];
  const float* cw     = (const float*)d_in[2];
  const float* cbias  = (const float*)d_in[3];
  const float* dtbias = (const float*)d_in[4];
  const float* A_log  = (const float*)d_in[5];
  const float* nw     = (const float*)d_in[6];
  const float* opw    = (const float*)d_in[7];
  float* out = (float*)d_out;

  float* ws = (float*)d_ws;
  size_t o = 0;
  auto take = [&](size_t nfloats) { float* p = ws + o; o += (nfloats + 63) & ~(size_t)63; return p; };
  const int M = NBATCH * LB;  // 8192

  float* zx   = take((size_t)M * NPROJ);              // in_proj output (f32)
  float* dtb  = take((size_t)M * 32);                 // dt per (b,l,h)
  float* ab   = take((size_t)M * 32);                 // a = dt*A
  float* Xdt  = take((size_t)M * DINNER);             // x * dt
  float* Bm   = take((size_t)M * 64);
  float* Cm   = take((size_t)M * 64);
  float* Acum = take((size_t)2048 * CHUNKL);          // (b,dir,c,h,l)
  float* tb   = take((size_t)2048);                   // chunk totals
  float* G    = take((size_t)64 * 65536);             // (b,dir,c)[256][256]
  float* st   = take((size_t)2048 * 4096);            // states (b,dir,c,h)[p][n]
  float* pv   = take((size_t)2048 * 4096);            // prev states
  float* yf   = take((size_t)M * DINNER);
  float* yb   = take((size_t)M * DINNER);
  bf16* ubf   = (bf16*)take((size_t)M * 1024 / 2);    // bf16 copies (2 per float slot)
  bf16* ipwbf = (bf16*)take((size_t)NPROJ * 1024 / 2);
  bf16* opwbf = (bf16*)take((size_t)1024 * DINNER / 2);
  bf16* ynbf  = (bf16*)take((size_t)M * DINNER / 2);

  // one-time bf16 conversions
  int nu4 = M * 1024 / 4;
  cvt_bf16_kernel<<<(nu4 + 255) / 256, 256, 0, stream>>>(u, ubf, nu4);
  int nw4 = NPROJ * 1024 / 4;
  cvt_bf16_kernel<<<(nw4 + 255) / 256, 256, 0, stream>>>(ipw, ipwbf, nw4);
  int no4 = 1024 * DINNER / 4;
  cvt_bf16_kernel<<<(no4 + 255) / 256, 256, 0, stream>>>(opw, opwbf, no4);

  // in_proj
  gemm_bf16_kernel<<<dim3(M / 128, (NPROJ + 255) / 256), 256, 0, stream>>>(
      ubf, ipwbf, zx, M, NPROJ, 1024);

  int dtTotal = M * 32;
  dt_kernel<<<(dtTotal + 255) / 256, 256, 0, stream>>>(zx, dtbias, A_log, dtb, ab, dtTotal);

  int convTotal = M * CONVDIM;
  conv_kernel<<<(convTotal + 255) / 256, 256, 0, stream>>>(zx, cw, cbias, dtb, Xdt, Bm, Cm, convTotal);

  cumsum_kernel<<<8, 256, 0, stream>>>(ab, Acum, tb);

  ssd_g_kernel<<<dim3(4, 4, 64), 128, 0, stream>>>(Bm, Cm, G);

  ssd_states_kernel<<<2048, 128, 0, stream>>>(Xdt, Bm, Acum, tb, st);

  scan_kernel<<<128, 256, 0, stream>>>(st, tb, pv);

  ssd_y_kernel<<<2048, 256, 0, stream>>>(G, Xdt, Cm, pv, Acum, yf, yb);

  norm_kernel<<<M, 256, 0, stream>>>(yf, yb, Xdt, Bm, Cm, zx, nw, ynbf);

  // out_proj
  gemm_bf16_kernel<<<dim3(M / 128, 1024 / 256), 256, 0, stream>>>(
      ynbf, opwbf, out, M, 1024, DINNER);
}
